// NodeModel_19894288515268
// MI455X (gfx1250) — matrix operand order repaired and checked by reading, not compiled, not run on hardware
//
#include <hip/hip_runtime.h>

typedef __attribute__((ext_vector_type(16))) _Float16 v16h;
typedef __attribute__((ext_vector_type(8)))  float    v8f;
typedef __attribute__((ext_vector_type(4)))  float    v4f;

#define F_X   8
#define F_E   8
#define MSG   16   // F_X + F_E
#define H_DIM 25   // F_E + 2*F_X + F_U

// ---------------- zero workspace (summed + cnt) ----------------
__global__ void zero_ws_kernel(float* __restrict__ ws, long long count) {
  long long i = (long long)blockIdx.x * blockDim.x + threadIdx.x;
  long long stride = (long long)gridDim.x * blockDim.x;
  for (; i < count; i += stride) ws[i] = 0.0f;
}

// ---------------- edge phase: gather + scatter-add ----------------
__global__ void edge_kernel(const float* __restrict__ x,
                            const long long* __restrict__ ei,
                            const float* __restrict__ ea,
                            float* __restrict__ summed,
                            float* __restrict__ cnt,
                            long long n_edges) {
  long long e = (long long)blockIdx.x * blockDim.x + threadIdx.x;
  if (e >= n_edges) return;
  // streamed once: non-temporal so it does not evict L2-resident x/accumulators
  long long row = __builtin_nontemporal_load(ei + e);
  long long col = __builtin_nontemporal_load(ei + n_edges + e);
  const v4f* xr = (const v4f*)(x + row * F_X);        // 32B-aligned
  v4f x0 = xr[0];
  v4f x1 = xr[1];
  const v4f* er = (const v4f*)(ea + e * F_E);
  v4f e0 = __builtin_nontemporal_load(er + 0);
  v4f e1 = __builtin_nontemporal_load(er + 1);

  float msg[MSG] = {x0.x, x0.y, x0.z, x0.w, x1.x, x1.y, x1.z, x1.w,
                    e0.x, e0.y, e0.z, e0.w, e1.x, e1.y, e1.z, e1.w};
  float* dst = summed + col * MSG;
#pragma unroll
  for (int j = 0; j < MSG; ++j)
    __hip_atomic_fetch_add(dst + j, msg[j], __ATOMIC_RELAXED, __HIP_MEMORY_SCOPE_AGENT);
  __hip_atomic_fetch_add(cnt + col, 1.0f, __ATOMIC_RELAXED, __HIP_MEMORY_SCOPE_AGENT);
}

// ---------------- node MLP via WMMA ----------------
// One wave per 16-node tile. K padded 25 -> 32.
// Layer1: D0 = A(16x32) x W1cols[0:16], D1 = A x W1cols[16:32]  (2 WMMAs)
// Layer2: D2 = H(16x32) x W2cols[0:16]                          (1 WMMA)
__global__ void __launch_bounds__(256)
mlp_kernel(const float* __restrict__ x,
           const float* __restrict__ summed,
           const float* __restrict__ cnt,
           const float* __restrict__ u,
           const float* __restrict__ W1,
           const float* __restrict__ b1,
           const float* __restrict__ W2,
           const float* __restrict__ b2,
           float* __restrict__ out,
           int n_nodes, int num_tiles) {
  __shared__ _Float16 hbuf[8 * 16 * 32];   // per-wave 16x32 f16 hidden tile

  const int lane = threadIdx.x & 31;
  const int wave = threadIdx.x >> 5;
  const int tile = blockIdx.x * 8 + wave;
  const bool active = tile < num_tiles;

  const int m  = lane & 15;   // row (A) or col (B/C)
  const int hi = lane >> 4;   // half-wave select
  _Float16* myh = hbuf + wave * (16 * 32);

  if (active) {
    const int base = tile * 16;
    const int node = base + m;
    const bool valid = node < n_nodes;
    const float uval = u[0];

    // A tile: lane holds row M=m, elems i: K = (i/8)*16 + hi*8 + (i%8)
    v16h a;
#pragma unroll
    for (int i = 0; i < 16; ++i) {
      const int K = ((i >> 3) << 4) + (hi << 3) + (i & 7);
      float v = 0.0f;
      if (valid) {
        if (K < F_X) {
          v = x[node * F_X + K];
        } else if (K < F_X + MSG) {
          const int j = K - F_X;
          const float c = cnt[node];
          v = summed[node * MSG + j] / fmaxf(c, 1.0f);
        } else if (K == F_X + MSG) {   // K == 24 -> u
          v = uval;
        }
      }
      a[i] = (_Float16)v;
    }

    // B tiles of W1 (row-major [25][25]): lane holds col N, elems i: K = (i/8)*16 + hi*8 + (i%8)
    const int N0 = m, N1 = 16 + m;
    v16h bt0, bt1;
#pragma unroll
    for (int i = 0; i < 16; ++i) {
      const int K = ((i >> 3) << 4) + (hi << 3) + (i & 7);
      bt0[i] = (_Float16)((K < H_DIM) ? W1[K * H_DIM + N0] : 0.0f);
      bt1[i] = (_Float16)((K < H_DIM && N1 < H_DIM) ? W1[K * H_DIM + N1] : 0.0f);
    }

    v8f c0 = {}, c1 = {};
    c0 = __builtin_amdgcn_wmma_f32_16x16x32_f16(false, a, false, bt0, (short)0, c0, false, false);
    c1 = __builtin_amdgcn_wmma_f32_16x16x32_f16(false, a, false, bt1, (short)0, c1, false, false);

    // bias + relu; stash hidden 16x32 tile to LDS (cols >= 25 zeroed)
    const float bia0 = b1[N0];
    const float bia1 = (N1 < H_DIM) ? b1[N1] : 0.0f;
#pragma unroll
    for (int j = 0; j < 8; ++j) {
      const int M = j + (hi << 3);
      const float h0 = fmaxf(c0[j] + bia0, 0.0f);
      const float h1 = (N1 < H_DIM) ? fmaxf(c1[j] + bia1, 0.0f) : 0.0f;
      myh[M * 32 + N0] = (_Float16)h0;
      myh[M * 32 + N1] = (_Float16)h1;
    }
  }

  __syncthreads();   // cross-lane D-layout -> A-layout exchange through LDS

  if (active) {
    const int base = tile * 16;

    // A2 tile from LDS hidden
    v16h a2;
#pragma unroll
    for (int i = 0; i < 16; ++i) {
      const int K = ((i >> 3) << 4) + (hi << 3) + (i & 7);
      a2[i] = myh[m * 32 + K];
    }
    // B2 tile of W2 (row-major [25][8]), cols 8..15 zero
    v16h bt2;
#pragma unroll
    for (int i = 0; i < 16; ++i) {
      const int K = ((i >> 3) << 4) + (hi << 3) + (i & 7);
      bt2[i] = (_Float16)((K < H_DIM && m < F_X) ? W2[K * F_X + m] : 0.0f);
    }

    v8f c2 = {};
    c2 = __builtin_amdgcn_wmma_f32_16x16x32_f16(false, a2, false, bt2, (short)0, c2, false, false);

    if (m < F_X) {
      const float bia = b2[m];
#pragma unroll
      for (int j = 0; j < 8; ++j) {
        const int M = j + (hi << 3);
        const int node = base + M;
        if (node < n_nodes) out[node * F_X + m] = c2[j] + bia;
      }
    }
  }
}

extern "C" void kernel_launch(void* const* d_in, const int* in_sizes, int n_in,
                              void* d_out, int out_size, void* d_ws, size_t ws_size,
                              hipStream_t stream) {
  const float*     x  = (const float*)d_in[0];
  const long long* ei = (const long long*)d_in[1];   // int64 edge_index [2, E]
  const float*     ea = (const float*)d_in[2];
  const float*     u  = (const float*)d_in[3];
  // d_in[4] = batch (unused: single graph, u broadcast)
  const float*     W1 = (const float*)d_in[5];
  const float*     b1 = (const float*)d_in[6];
  const float*     W2 = (const float*)d_in[7];
  const float*     b2 = (const float*)d_in[8];
  float* out = (float*)d_out;

  const int       n_nodes = in_sizes[0] / F_X;
  const long long n_edges = (long long)in_sizes[2] / F_E;

  float* summed = (float*)d_ws;                       // [n_nodes, 16]
  float* cnt    = summed + (size_t)n_nodes * MSG;     // [n_nodes]

  const long long zcount = (long long)n_nodes * (MSG + 1);
  zero_ws_kernel<<<2048, 256, 0, stream>>>(summed, zcount);

  const long long eblocks = (n_edges + 255) / 256;
  edge_kernel<<<dim3((unsigned)eblocks), 256, 0, stream>>>(x, ei, ea, summed, cnt, n_edges);

  const int num_tiles = (n_nodes + 15) / 16;
  const int mblocks   = (num_tiles + 7) / 8;
  mlp_kernel<<<mblocks, 256, 0, stream>>>(x, summed, cnt, u, W1, b1, W2, b2, out,
                                          n_nodes, num_tiles);
}
